// AttentionHead_51496657879663
// MI455X (gfx1250) — compile-verified, matrix operand order checked by reading, not run on hardware
//
#include <hip/hip_runtime.h>

#define BATCH 8
#define SEQ   2048
#define DIM   1024

typedef __attribute__((ext_vector_type(16))) __bf16 v16bf;
typedef __attribute__((ext_vector_type(8)))  float  v8f;
typedef __attribute__((ext_vector_type(2)))  __bf16 v2bf;

union Frag16 { v16bf v; uint4 q[2]; };

#if defined(__has_builtin)
#if __has_builtin(__builtin_amdgcn_cvt_pk_bf16_f32)
#define HAVE_PK_BF16 1
#endif
#endif

// pack two floats into two bf16 in one dword (hardware v_cvt_pk_bf16_f32)
__device__ inline unsigned pack2bf(float lo, float hi) {
#ifdef HAVE_PK_BF16
    union { v2bf v; unsigned u; } c;
    c.v = __builtin_amdgcn_cvt_pk_bf16_f32(lo, hi);
    return c.u;
#else
    unsigned r;
    asm("v_cvt_pk_bf16_f32 %0, %1, %2" : "=v"(r) : "v"(lo), "v"(hi));
    return r;
#endif
}

// Load a 16x32 bf16 A-fragment (or 32x16 B-fragment from k-major storage).
// ISA layout: lane L (<16): row m=L, K runs {0..7},{16..23}; lane L+16: row m=L, K runs {8..15},{24..31}.
__device__ inline v16bf load_frag(const unsigned short* base, int row0, int ld, int k0) {
    const int lane = threadIdx.x & 31;
    const int m    = lane & 15;
    const int koff = (lane & 16) ? 8 : 0;
    const unsigned short* p = base + (size_t)(row0 + m) * ld + k0 + koff;
    Frag16 f;
    f.q[0] = *(const uint4*)(p);        // K = koff .. koff+7   (16B)
    f.q[1] = *(const uint4*)(p + 16);   // K = koff+16 .. +23   (16B)
    return f.v;
}

// ---------------------------------------------------------------------------
// Kernel 1: q/k/v projections. C[row, n] = sum_f x[row,f] * W[n,f]
// z==0 -> q (row-major bf16), z==1 -> k (row-major bf16), z==2 -> v transposed vT[b][d][s].
// Ping-pong LDS double buffering: one barrier per K-step, next tile's global
// loads in flight during the WMMAs.
// ---------------------------------------------------------------------------
__global__ __launch_bounds__(256) void qkv_kernel(
    const float* __restrict__ x,
    const float* __restrict__ Wq, const float* __restrict__ Wk, const float* __restrict__ Wv,
    unsigned short* __restrict__ qo, unsigned short* __restrict__ ko, unsigned short* __restrict__ vT)
{
    __shared__ unsigned short As[2][128 * 40];   // 128 rows x 32 K (+8 pad), bf16
    __shared__ unsigned short Bs[2][128 * 40];

    const int t    = threadIdx.x;
    const int row0 = blockIdx.x * 128;   // global (b*S+s) row; tile lies in one batch
    const int n0   = blockIdx.y * 128;   // output feature
    const int z    = blockIdx.z;
    const float* W = (z == 0) ? Wq : (z == 1) ? Wk : Wv;

    const int r_ld = t >> 3;             // 0..31 base row for loads
    const int c4   = t & 7;              // float4 column within 32-wide K tile

    const int wid  = t >> 5;
    const int Moff = (wid >> 1) * 32;    // 4 M-slices of 32
    const int Noff = (wid & 1) * 64;     // 2 N-slices of 64

    v8f acc[2][4];
#pragma unroll
    for (int i = 0; i < 2; ++i)
#pragma unroll
        for (int j = 0; j < 4; ++j) acc[i][j] = (v8f){0.f,0.f,0.f,0.f,0.f,0.f,0.f,0.f};

    float4 ra[4], rb[4];

    // prologue: fetch tile 0 into regs, store to buffer 0
#pragma unroll
    for (int i = 0; i < 4; ++i) {
        int r = r_ld + i * 32;
        ra[i] = *(const float4*)(x + (size_t)(row0 + r) * DIM + c4 * 4);
        rb[i] = *(const float4*)(W + (size_t)(n0 + r) * DIM + c4 * 4);
    }
#pragma unroll
    for (int i = 0; i < 4; ++i) {
        int r = r_ld + i * 32;
        uint2 ua = { pack2bf(ra[i].x, ra[i].y), pack2bf(ra[i].z, ra[i].w) };
        uint2 ub = { pack2bf(rb[i].x, rb[i].y), pack2bf(rb[i].z, rb[i].w) };
        *(uint2*)(As[0] + r * 40 + c4 * 4) = ua;
        *(uint2*)(Bs[0] + r * 40 + c4 * 4) = ub;
    }
    __syncthreads();

    for (int ks = 0; ks < DIM / 32; ++ks) {
        const int cur = ks & 1;
        if (ks + 1 < DIM / 32) {
            const int k0n = (ks + 1) * 32;
#pragma unroll
            for (int i = 0; i < 4; ++i) {
                int r = r_ld + i * 32;
                ra[i] = *(const float4*)(x + (size_t)(row0 + r) * DIM + k0n + c4 * 4);
                rb[i] = *(const float4*)(W + (size_t)(n0 + r) * DIM + k0n + c4 * 4);
            }
            if (ks + 2 < DIM / 32) {    // warm L2 for the tile after next
                const int k0p = (ks + 2) * 32;
                __builtin_prefetch(x + (size_t)(row0 + r_ld) * DIM + k0p + c4 * 4, 0, 1);
                __builtin_prefetch(W + (size_t)(n0 + r_ld) * DIM + k0p + c4 * 4, 0, 1);
            }
        }

        v16bf a0 = load_frag(As[cur], Moff,      40, 0);
        v16bf a1 = load_frag(As[cur], Moff + 16, 40, 0);
#pragma unroll
        for (int nf = 0; nf < 4; ++nf) {
            v16bf bb = load_frag(Bs[cur], Noff + nf * 16, 40, 0);
            acc[0][nf] = __builtin_amdgcn_wmma_f32_16x16x32_bf16(false, a0, false, bb, (short)0, acc[0][nf], false, false);
            acc[1][nf] = __builtin_amdgcn_wmma_f32_16x16x32_bf16(false, a1, false, bb, (short)0, acc[1][nf], false, false);
        }

        if (ks + 1 < DIM / 32) {
#pragma unroll
            for (int i = 0; i < 4; ++i) {
                int r = r_ld + i * 32;
                uint2 ua = { pack2bf(ra[i].x, ra[i].y), pack2bf(ra[i].z, ra[i].w) };
                uint2 ub = { pack2bf(rb[i].x, rb[i].y), pack2bf(rb[i].z, rb[i].w) };
                *(uint2*)(As[cur ^ 1] + r * 40 + c4 * 4) = ua;
                *(uint2*)(Bs[cur ^ 1] + r * 40 + c4 * 4) = ub;
            }
        }
        __syncthreads();
    }

    // epilogue: mode branch hoisted OUTSIDE the store loops
    const int lane = t & 31;
    const int half = (lane & 16) ? 8 : 0;
    const int nl   = lane & 15;
    if (z < 2) {
        unsigned short* out = (z == 0) ? qo : ko;
#pragma unroll
        for (int mi = 0; mi < 2; ++mi)
#pragma unroll
            for (int nf = 0; nf < 4; ++nf) {
                int col = n0 + Noff + nf * 16 + nl;
                unsigned short* orow =
                    out + (size_t)(row0 + Moff + mi * 16 + half) * DIM + col;
#pragma unroll
                for (int j = 0; j < 8; j += 2) {   // pair-convert, store lo/hi halves
                    unsigned p = pack2bf(acc[mi][nf][j], acc[mi][nf][j + 1]);
                    orow[(size_t)j * DIM]       = (unsigned short)(p & 0xffffu);
                    orow[(size_t)(j + 1) * DIM] = (unsigned short)(p >> 16);
                }
            }
    } else {
        const int bidx  = row0 >> 11;          // tile fully inside one batch
        const int sbase = (row0 & (SEQ - 1)) + Moff + half;
#pragma unroll
        for (int mi = 0; mi < 2; ++mi)
#pragma unroll
            for (int nf = 0; nf < 4; ++nf) {
                int col = n0 + Noff + nf * 16 + nl;
                // 8 consecutive s per lane -> one 16B store of 4 packed dwords
                uint4 pk = { pack2bf(acc[mi][nf][0], acc[mi][nf][1]),
                             pack2bf(acc[mi][nf][2], acc[mi][nf][3]),
                             pack2bf(acc[mi][nf][4], acc[mi][nf][5]),
                             pack2bf(acc[mi][nf][6], acc[mi][nf][7]) };
                *(uint4*)(vT + ((size_t)bidx * DIM + col) * SEQ + sbase + mi * 16) = pk;
            }
    }
}

// ---------------------------------------------------------------------------
// Kernel 2: fused scores + bias + softmax + P@V for a block of 32 q rows.
// 512 threads = 16 waves.
// Phase 1: wave w owns score cols [w*128, w*128+128) for both 16-row M-frags.
// Phase 2: wave w owns output cols [w*64, w*64+64).
// q block staged via ASYNC global->LDS DMA into a region later reused for P
// (Qs is dead before the first P write; barriers separate the lifetimes).
// ---------------------------------------------------------------------------
#define MQ    32
#define Q_LD  1032
#define P_LD  2056
__global__ __launch_bounds__(512) void attn_kernel(
    const unsigned short* __restrict__ q,
    const unsigned short* __restrict__ k,
    const unsigned short* __restrict__ vT,
    const float* __restrict__ emb,
    float* __restrict__ out)
{
    extern __shared__ char smem[];
    unsigned short* P  = (unsigned short*)smem;                  // 32 x 2056 bf16 (phase 2)
    unsigned short* Qs = (unsigned short*)smem;                  // 32 x 1032 bf16 (phase 1, aliased)
    float* red         = (float*)(smem + (size_t)MQ * P_LD * 2); // 16 waves x 32 rows

    const int t    = threadIdx.x;
    const int lane = t & 31;
    const int wid  = t >> 5;                 // 0..15
    const int b    = blockIdx.y;
    const int q0   = blockIdx.x * MQ;

    const unsigned short* qbase = q  + ((size_t)b * SEQ + q0) * DIM;
    const unsigned short* kbase = k  + (size_t)b * SEQ * DIM;
    const unsigned short* vbase = vT + (size_t)b * DIM * SEQ;

    const int half = (lane & 16) ? 8 : 0;
    const int nl   = lane & 15;

    // stage the 32x1024 bf16 q block into LDS with async DMA:
    // 4096 16B chunks / 512 threads = 8 async loads per lane (ASYNCcnt-tracked).
#pragma unroll
    for (int i = 0; i < 8; ++i) {
        int idx = t + i * 512;
        int r = idx >> 7, c = idx & 127;
        unsigned ldsa = (unsigned)(uintptr_t)(Qs + (size_t)r * Q_LD + c * 8);
        const unsigned short* g = qbase + (size_t)r * DIM + c * 8;
        asm volatile("global_load_async_to_lds_b128 %0, %1, off"
                     :: "v"(ldsa), "v"(g) : "memory");
    }
    asm volatile("s_wait_asynccnt 0x0" ::: "memory");
    __syncthreads();

    // ---- Phase 1: scores = q @ k^T over this wave's 128 columns, 32 rows ----
    const int n0 = wid * 128;
    v8f acc[2][8];
#pragma unroll
    for (int mi = 0; mi < 2; ++mi)
#pragma unroll
        for (int i = 0; i < 8; ++i) acc[mi][i] = (v8f){0.f,0.f,0.f,0.f,0.f,0.f,0.f,0.f};

    for (int k0 = 0; k0 < DIM; k0 += 32) {
        v16bf a0 = load_frag(Qs, 0,  Q_LD, k0);
        v16bf a1 = load_frag(Qs, 16, Q_LD, k0);
#pragma unroll
        for (int nf = 0; nf < 8; ++nf) {
            v16bf bb = load_frag(kbase, n0 + nf * 16, DIM, k0);
            acc[0][nf] = __builtin_amdgcn_wmma_f32_16x16x32_bf16(false, a0, false, bb, (short)0, acc[0][nf], false, false);
            acc[1][nf] = __builtin_amdgcn_wmma_f32_16x16x32_bf16(false, a1, false, bb, (short)0, acc[1][nf], false, false);
        }
    }

    // scale + relative-position bias, track per-row max
    float mymax[2][8];
#pragma unroll
    for (int mi = 0; mi < 2; ++mi)
#pragma unroll
        for (int j = 0; j < 8; ++j) mymax[mi][j] = -3.0e38f;
#pragma unroll
    for (int mi = 0; mi < 2; ++mi)
#pragma unroll
        for (int nf = 0; nf < 8; ++nf) {
            int kg = n0 + nf * 16 + nl;
#pragma unroll
            for (int j = 0; j < 8; ++j) {
                int qg = q0 + mi * 16 + half + j;
                int di = qg - kg; di = di < 0 ? -di : di;
                float s = acc[mi][nf][j] * 0.03125f + emb[di];   // 1/sqrt(1024)
                acc[mi][nf][j] = s;
                mymax[mi][j] = fmaxf(mymax[mi][j], s);
            }
        }
    // reduce across the 16 lanes sharing each row (stays within half)
#pragma unroll
    for (int mi = 0; mi < 2; ++mi)
#pragma unroll
        for (int j = 0; j < 8; ++j) {
            float v = mymax[mi][j];
            v = fmaxf(v, __shfl_xor(v, 1, 32));
            v = fmaxf(v, __shfl_xor(v, 2, 32));
            v = fmaxf(v, __shfl_xor(v, 4, 32));
            v = fmaxf(v, __shfl_xor(v, 8, 32));
            mymax[mi][j] = v;
        }
    if (nl == 0) {
#pragma unroll
        for (int mi = 0; mi < 2; ++mi)
#pragma unroll
            for (int j = 0; j < 8; ++j) red[wid * MQ + mi * 16 + half + j] = mymax[mi][j];
    }
    __syncthreads();
    float gmax[2][8];
#pragma unroll
    for (int mi = 0; mi < 2; ++mi)
#pragma unroll
        for (int j = 0; j < 8; ++j) {
            float v = -3.0e38f;
#pragma unroll
            for (int w = 0; w < 16; ++w) v = fmaxf(v, red[w * MQ + mi * 16 + half + j]);
            gmax[mi][j] = v;
        }
    __syncthreads();   // before red is reused for the sums

    // exp + per-row sum
    float mysum[2][8];
#pragma unroll
    for (int mi = 0; mi < 2; ++mi)
#pragma unroll
        for (int j = 0; j < 8; ++j) mysum[mi][j] = 0.f;
#pragma unroll
    for (int mi = 0; mi < 2; ++mi)
#pragma unroll
        for (int nf = 0; nf < 8; ++nf)
#pragma unroll
            for (int j = 0; j < 8; ++j) {
                float p = exp2f((acc[mi][nf][j] - gmax[mi][j]) * 1.4426950408889634f);
                acc[mi][nf][j] = p;
                mysum[mi][j] += p;
            }
#pragma unroll
    for (int mi = 0; mi < 2; ++mi)
#pragma unroll
        for (int j = 0; j < 8; ++j) {
            float v = mysum[mi][j];
            v += __shfl_xor(v, 1, 32);
            v += __shfl_xor(v, 2, 32);
            v += __shfl_xor(v, 4, 32);
            v += __shfl_xor(v, 8, 32);
            mysum[mi][j] = v;
        }
    if (nl == 0) {
#pragma unroll
        for (int mi = 0; mi < 2; ++mi)
#pragma unroll
            for (int j = 0; j < 8; ++j) red[wid * MQ + mi * 16 + half + j] = mysum[mi][j];
    }
    __syncthreads();
    float rinv[2][8];
#pragma unroll
    for (int mi = 0; mi < 2; ++mi)
#pragma unroll
        for (int j = 0; j < 8; ++j) {
            float v = 0.f;
#pragma unroll
            for (int w = 0; w < 16; ++w) v += red[w * MQ + mi * 16 + half + j];
            rinv[mi][j] = 1.0f / v;
        }
    __syncthreads();   // all Qs reads done; region becomes P

    // write normalized probabilities to LDS (bf16, pair-convert, lo/hi b16 stores)
#pragma unroll
    for (int mi = 0; mi < 2; ++mi)
#pragma unroll
        for (int nf = 0; nf < 8; ++nf) {
            unsigned short* prow = P + (size_t)(mi * 16 + half) * P_LD + n0 + nf * 16 + nl;
#pragma unroll
            for (int j = 0; j < 8; j += 2) {
                unsigned p = pack2bf(acc[mi][nf][j]     * rinv[mi][j],
                                     acc[mi][nf][j + 1] * rinv[mi][j + 1]);
                prow[(size_t)j * P_LD]       = (unsigned short)(p & 0xffffu);
                prow[(size_t)(j + 1) * P_LD] = (unsigned short)(p >> 16);
            }
        }
    __syncthreads();

    // ---- Phase 2: out = P @ V  (B-fragments from vT: contiguous in s) ----
    const int on0 = wid * 64;
    v8f oacc[2][4];
#pragma unroll
    for (int mi = 0; mi < 2; ++mi)
#pragma unroll
        for (int i = 0; i < 4; ++i) oacc[mi][i] = (v8f){0.f,0.f,0.f,0.f,0.f,0.f,0.f,0.f};

    for (int k0 = 0; k0 < SEQ; k0 += 32) {
        v16bf a0 = load_frag(P, 0,  P_LD, k0);
        v16bf a1 = load_frag(P, 16, P_LD, k0);
#pragma unroll
        for (int nf = 0; nf < 4; ++nf) {
            v16bf bb = load_frag(vbase, on0 + nf * 16, SEQ, k0);
            oacc[0][nf] = __builtin_amdgcn_wmma_f32_16x16x32_bf16(false, a0, false, bb, (short)0, oacc[0][nf], false, false);
            oacc[1][nf] = __builtin_amdgcn_wmma_f32_16x16x32_bf16(false, a1, false, bb, (short)0, oacc[1][nf], false, false);
        }
    }

    float* obase = out + ((size_t)b * SEQ + q0) * DIM;
#pragma unroll
    for (int mi = 0; mi < 2; ++mi)
#pragma unroll
        for (int nf = 0; nf < 4; ++nf)
#pragma unroll
            for (int j = 0; j < 8; ++j)
                obase[(size_t)(mi * 16 + half + j) * DIM + on0 + nf * 16 + nl] = oacc[mi][nf][j];
}

// ---------------------------------------------------------------------------
extern "C" void kernel_launch(void* const* d_in, const int* in_sizes, int n_in,
                              void* d_out, int out_size, void* d_ws, size_t ws_size,
                              hipStream_t stream)
{
    const float* x   = (const float*)d_in[0];
    const float* Wq  = (const float*)d_in[1];
    const float* Wk  = (const float*)d_in[2];
    const float* Wv  = (const float*)d_in[3];
    const float* emb = (const float*)d_in[4];
    float* out = (float*)d_out;

    const size_t elems = (size_t)BATCH * SEQ * DIM;
    unsigned short* qb = (unsigned short*)d_ws;     // bf16 q   (32 MB)
    unsigned short* kb = qb + elems;                // bf16 k   (32 MB)
    unsigned short* vT = kb + elems;                // bf16 v^T (32 MB)

    dim3 g1(BATCH * SEQ / 128, DIM / 128, 3);
    qkv_kernel<<<g1, 256, 0, stream>>>(x, Wq, Wk, Wv, qb, kb, vT);

    dim3 g2(SEQ / MQ, BATCH);
    size_t shmem = (size_t)MQ * P_LD * sizeof(unsigned short)   // P (aliases Qs)
                 + 16 * MQ * sizeof(float);                     // red
    attn_kernel<<<g2, 512, shmem, stream>>>(qb, kb, vT, emb, out);
}
